// Performer_55722905698489
// MI455X (gfx1250) — matrix operand order repaired and checked by reading, not compiled, not run on hardware
//
#include <hip/hip_runtime.h>
#include <hip/hip_bf16.h>

// ---- Problem constants (fixed by the reference) ----
#define B_      4
#define N_      8192
#define DIM_    512
#define HEADS_  8
#define DH_     64
#define NBF_    256
#define HID_    2048
#define NT_     (B_*N_)        // 32768 rows
#define BH_     (B_*HEADS_)    // 32 head-batches
#define KSPLIT_ 16             // deterministic split-K for ctx GEMM
#define LDSB_STRIDE 40         // padded row stride (halves): 80B = 20-bank stride

typedef __attribute__((ext_vector_type(16))) _Float16 v16h;
typedef __attribute__((ext_vector_type(8)))  _Float16 v8h;
typedef __attribute__((ext_vector_type(8)))  float    v8f;

// ---------- helpers ----------
__device__ inline unsigned f2ord(float f) {
    unsigned u = __float_as_uint(f);
    return (u & 0x80000000u) ? ~u : (u | 0x80000000u);
}
__device__ inline float ord2f(unsigned u) {
    return __uint_as_float((u & 0x80000000u) ? (u & 0x7fffffffu) : ~u);
}

// =====================================================================
// Templated WMMA GEMM:  C[M,N] = sum_k A[m,k] * BT[n,k]  (+ epilogue)
//   AMODE 0: A row-major, stride lda (batched by zb)
//   AMODE 1: A is o16 in [B,H,N,DH] head-blocked layout, addressed by (r,k)
//   EPI 0: f32 store (+bias)(+resid)     EPI 1: f16 store (+bias)(gelu/rscale)
//   EPI 2: f16 head-blocked transpose store -> [B,H,DH,N] (+bias)
//   EPI 3: f32 split-K partial store indexed by full blockIdx.z (no bias)
// B tile (64 cols x 32 k) is staged to LDS with async-DMA
// (global_load_async_to_lds_b128, ASYNCcnt), double-buffered; A fragment is
// register-double-buffered; all 4 B fragments are read from LDS before the
// 4 WMMAs so only partial s_wait_dscnt waits are needed.
// =====================================================================
template<int AMODE, int EPI, bool BIAS, bool GELU, bool RSCALE, bool RESID>
__global__ __launch_bounds__(256) void gemm_wmma(
    const _Float16* __restrict__ A, long lda, long batchA,
    const _Float16* __restrict__ Bt, long ldb, long batchB,
    const float* __restrict__ bias,
    const float* __restrict__ rscale, long batchS,
    const float* __restrict__ resid,
    void* __restrict__ outp, long ldc, long batchC,
    int Ktot, int ksplit)
{
    __shared__ __align__(16) _Float16 ldsB[2][64 * LDSB_STRIDE];

    const int tid  = threadIdx.x;
    const int lane = tid & 31;
    const int wv   = tid >> 5;
    const int hlf  = lane >> 4;   // lane half selects K sub-block (wave32 layout)
    const int l15  = lane & 15;

    const int z  = blockIdx.z;
    const int zb = z / ksplit;
    const int zk = z - zb * ksplit;
    const int kLen = Ktot / ksplit;
    const int k0 = zk * kLen, k1 = k0 + kLen;

    const long rowBase = (long)blockIdx.x * 128 + (long)wv * 16;
    const long mrow    = rowBase + l15;          // A row held by this lane
    const long colBase = (long)blockIdx.y * 64;

    const _Float16* Ab = A  + (long)zb * batchA;
    const _Float16* Bb = Bt + (long)zb * batchB;

    // ---- async B-tile staging: thread j copies one b128 (row j>>2, seg j&3)
    const int srow = tid >> 2;
    const int sseg = (tid & 3) * 8;
    auto stageB = [&](int kc, int buf) {
        const _Float16* g = Bb + (colBase + srow) * ldb + kc + sseg;
        const unsigned loff = (unsigned)(size_t)(&ldsB[buf][srow * LDSB_STRIDE + sseg]);
        asm volatile("global_load_async_to_lds_b128 %0, %1, off"
                     :: "v"(loff), "v"((unsigned long long)(size_t)g)
                     : "memory");
    };

    // ---- A fragment load (16x32 f16): VGPR0-3 = K[8*hlf..+7], VGPR4-7 = K[16+8*hlf..+7]
    auto loadA = [&](int kc) -> v16h {
        const _Float16* pa;
        if constexpr (AMODE == 0) {
            pa = Ab + mrow * lda + kc + 8 * hlf;
        } else {
            // o16 layout: [b][h][n][d];  r = b*N_+n, k = h*DH_+d
            pa = Ab + ((mrow >> 13) * HEADS_ + (kc >> 6)) * (long)(N_ * DH_)
                    + (mrow & (N_ - 1)) * DH_ + (kc & (DH_ - 1)) + 8 * hlf;
        }
        v8h a0 = *(const v8h*)pa;
        v8h a1 = *(const v8h*)(pa + 16);
        __builtin_prefetch((const void*)(pa + 32), 0, 1);   // global_prefetch_b8
        return __builtin_shufflevector(a0, a1, 0,1,2,3,4,5,6,7,8,9,10,11,12,13,14,15);
    };

    v8f acc[4] = {};

    stageB(k0, 0);
    v16h av = loadA(k0);
    int buf = 0;
    for (int kc = k0; kc < k1; kc += 32, buf ^= 1) {
        // previous async stage of this buffer is complete + visible to all waves
        asm volatile("s_wait_asynccnt 0x0" ::: "memory");
        __syncthreads();
        if (kc + 32 < k1) stageB(kc + 32, buf ^ 1);

        v16h avn = av;
        if (kc + 32 < k1) avn = loadA(kc + 32);   // register double-buffer A

        // Read all 4 B fragments first (8x ds_load_b128), then issue the WMMAs
        // so the DS waits are partial rather than full drains.
        v16h bvs[4];
#pragma unroll
        for (int t = 0; t < 4; ++t) {
            // 16-bit B fragment 32x16: lane n=l15, K = kc + 8*hlf + [0..7], kc + 16 + 8*hlf + [0..7]
            const _Float16* pb = &ldsB[buf][(t * 16 + l15) * LDSB_STRIDE + 8 * hlf];
            v8h b0 = *(const v8h*)pb;
            v8h b1 = *(const v8h*)(pb + 16);
            bvs[t] = __builtin_shufflevector(b0, b1, 0,1,2,3,4,5,6,7,8,9,10,11,12,13,14,15);
        }
#pragma unroll
        for (int t = 0; t < 4; ++t) {
            acc[t] = __builtin_amdgcn_wmma_f32_16x16x32_f16(
                        false, av, false, bvs[t], (short)0, acc[t], false, false);
        }
        av = avn;
    }

    // Epilogue. C/D layout: lane n=l15, VGPR r holds M = r + 8*hlf.
#pragma unroll
    for (int t = 0; t < 4; ++t) {
        const long nc = colBase + t * 16 + l15;
        float bi = 0.f;
        if constexpr (BIAS) bi = bias[nc];
#pragma unroll
        for (int r = 0; r < 8; ++r) {
            const long mr = rowBase + r + 8 * hlf;
            if constexpr (EPI == 3) {
                ((float*)outp)[(long)z * batchC + mr * ldc + nc] = acc[t][r];
            } else {
                float v = acc[t][r] + bi;
                if constexpr (RSCALE) v *= rscale[(long)zb * batchS + mr];
                if constexpr (GELU)   v = 0.5f * v * (1.0f + erff(v * 0.70710678118654752f));
                if constexpr (EPI == 0) {
                    const long idx = (long)zb * batchC + mr * ldc + nc;
                    float o = v;
                    if constexpr (RESID) o += resid[idx];
                    ((float*)outp)[idx] = o;
                } else if constexpr (EPI == 1) {
                    ((_Float16*)outp)[(long)zb * batchC + mr * ldc + nc] = (_Float16)v;
                } else if constexpr (EPI == 2) {
                    // store transposed into [B][H][DH][N] : r=b*N+n col=h*DH+d
                    const long idx = ((mr >> 13) * HEADS_ + (nc >> 6)) * (long)(N_ * DH_)
                                   + (nc & (DH_ - 1)) * (long)N_ + (mr & (N_ - 1));
                    ((_Float16*)outp)[idx] = (_Float16)v;
                }
            }
        }
    }
}

// =====================================================================
// LayerNorm: one 256-thread block per row of [NT_, DIM_], writes f16.
// =====================================================================
__global__ __launch_bounds__(256) void ln_kernel(
    const float* __restrict__ x, const float* __restrict__ g,
    const float* __restrict__ b, _Float16* __restrict__ y)
{
    __shared__ float red[256];
    const int row = blockIdx.x, tid = threadIdx.x;
    const float* xr = x + (long)row * DIM_;
    float x0 = xr[tid], x1 = xr[tid + 256];
    red[tid] = x0 + x1;
    __syncthreads();
    for (int o = 128; o > 0; o >>= 1) { if (tid < o) red[tid] += red[tid + o]; __syncthreads(); }
    const float mu = red[0] * (1.0f / DIM_);
    __syncthreads();
    const float d0 = x0 - mu, d1 = x1 - mu;
    red[tid] = d0 * d0 + d1 * d1;
    __syncthreads();
    for (int o = 128; o > 0; o >>= 1) { if (tid < o) red[tid] += red[tid + o]; __syncthreads(); }
    const float rstd = rsqrtf(red[0] * (1.0f / DIM_) + 1e-5f);
    _Float16* yr = y + (long)row * DIM_;
    yr[tid]       = (_Float16)(d0 * rstd * g[tid]       + b[tid]);
    yr[tid + 256] = (_Float16)(d1 * rstd * g[tid + 256] + b[tid + 256]);
}

// =====================================================================
// FAVOR+ kernels. proj staged in LDS as f16 [d][m] (32 KB, conflict-free).
// =====================================================================
#define KROWS 64
__global__ __launch_bounds__(256) void kmax_kernel(
    const float* __restrict__ kf, const float* __restrict__ proj,
    unsigned* __restrict__ kmax)
{
    __shared__ _Float16 pl[DH_ * NBF_];
    __shared__ float krow[DH_];
    __shared__ float red[256];
    const int tid = threadIdx.x;
    const int bh = blockIdx.y, b = bh >> 3, h = bh & 7;
    for (int j = tid; j < DH_ * NBF_; j += 256) {
        int d = j >> 8, m = j & 255;
        pl[j] = (_Float16)proj[m * DH_ + d];
    }
    __syncthreads();
    const int n0 = blockIdx.x * KROWS, m = tid;
    float mx = -1e30f;
    for (int rr = 0; rr < KROWS; ++rr) {
        const int n = n0 + rr;
        if (tid < DH_) krow[tid] = kf[((long)(b * N_ + n)) * DIM_ + h * DH_ + tid];
        __syncthreads();
        float dot = 0.f;
#pragma unroll
        for (int d = 0; d < DH_; ++d) dot += krow[d] * (float)pl[d * NBF_ + m];
        mx = fmaxf(mx, dot * 0.35355339059327379f);
        __syncthreads();
    }
    red[tid] = mx;
    __syncthreads();
    for (int o = 128; o > 0; o >>= 1) { if (tid < o) red[tid] = fmaxf(red[tid], red[tid + o]); __syncthreads(); }
    if (tid == 0) atomicMax(kmax + bh, f2ord(red[0]));
}

__global__ __launch_bounds__(256) void kp_kernel(
    const float* __restrict__ kf, const float* __restrict__ proj,
    const unsigned* __restrict__ kmax, _Float16* __restrict__ kpT)
{
    __shared__ _Float16 pl[DH_ * NBF_];
    __shared__ float krow[DH_];
    const int tid = threadIdx.x;
    const int bh = blockIdx.y, b = bh >> 3, h = bh & 7;
    for (int j = tid; j < DH_ * NBF_; j += 256) {
        int d = j >> 8, m = j & 255;
        pl[j] = (_Float16)proj[m * DH_ + d];
    }
    const float Mh = ord2f(kmax[bh]);
    __syncthreads();
    const int n0 = blockIdx.x * KROWS, m = tid;
    for (int rr = 0; rr < KROWS; ++rr) {
        const int n = n0 + rr;
        if (tid < DH_) krow[tid] = kf[((long)(b * N_ + n)) * DIM_ + h * DH_ + tid];
        __syncthreads();
        float ss = 0.f, dot = 0.f;
#pragma unroll
        for (int d = 0; d < DH_; ++d) { float kv = krow[d]; ss += kv * kv; dot += kv * (float)pl[d * NBF_ + m]; }
        const float val = 0.0625f * (expf(dot * 0.35355339059327379f - 0.0625f * ss - Mh) + 1e-4f);
        kpT[((long)(bh * NBF_ + m)) * N_ + n] = (_Float16)val;   // transposed [B,H,M,N]
        __syncthreads();
    }
}

__global__ __launch_bounds__(256) void ksum_kernel(
    const _Float16* __restrict__ kpT, float* __restrict__ ksum)
{
    __shared__ float red[256];
    const long row = blockIdx.x;           // BH_*NBF_ rows
    float s = 0.f;
    for (int n = threadIdx.x; n < N_; n += 256) s += (float)kpT[row * N_ + n];
    red[threadIdx.x] = s;
    __syncthreads();
    for (int o = 128; o > 0; o >>= 1) { if (threadIdx.x < o) red[threadIdx.x] += red[threadIdx.x + o]; __syncthreads(); }
    if (threadIdx.x == 0) ksum[row] = red[0];
}

#define QROWS 32
__global__ __launch_bounds__(256) void qp_kernel(
    const float* __restrict__ qf, const float* __restrict__ proj,
    const float* __restrict__ ksum, _Float16* __restrict__ qp,
    float* __restrict__ dinv)
{
    __shared__ _Float16 pl[DH_ * NBF_];
    __shared__ float qrow[DH_];
    __shared__ float red[256];
    const int tid = threadIdx.x;
    const int bh = blockIdx.y, b = bh >> 3, h = bh & 7;
    for (int j = tid; j < DH_ * NBF_; j += 256) {
        int d = j >> 8, m = j & 255;
        pl[j] = (_Float16)proj[m * DH_ + d];
    }
    const float ks = ksum[bh * NBF_ + tid];
    __syncthreads();
    const int n0 = blockIdx.x * QROWS, m = tid;
    for (int rr = 0; rr < QROWS; ++rr) {
        const int n = n0 + rr;
        if (tid < DH_) qrow[tid] = qf[((long)(b * N_ + n)) * DIM_ + h * DH_ + tid];
        __syncthreads();
        float ss = 0.f, dot = 0.f;
#pragma unroll
        for (int d = 0; d < DH_; ++d) { float qv = qrow[d]; ss += qv * qv; dot += qv * (float)pl[d * NBF_ + m]; }
        const float dd = dot * 0.35355339059327379f;
        red[tid] = dd;
        __syncthreads();
        for (int o = 128; o > 0; o >>= 1) { if (tid < o) red[tid] = fmaxf(red[tid], red[tid + o]); __syncthreads(); }
        const float rmax = red[0];
        __syncthreads();
        const float val = 0.0625f * (expf(dd - 0.0625f * ss - rmax) + 1e-4f);
        qp[((long)(bh * N_ + n)) * NBF_ + m] = (_Float16)val;
        red[tid] = val * ks;
        __syncthreads();
        for (int o = 128; o > 0; o >>= 1) { if (tid < o) red[tid] += red[tid + o]; __syncthreads(); }
        if (tid == 0) dinv[(long)bh * N_ + n] = 1.0f / red[0];
        __syncthreads();
    }
}

// Deterministic split-K reduction for ctx: ctx[j] = sum_zk part[bh*16+zk][j']
__global__ __launch_bounds__(256) void ctx_reduce_kernel(
    const float* __restrict__ part, float* __restrict__ ctx)
{
    const long j = (long)blockIdx.x * 256 + threadIdx.x;   // BH_*NBF_*DH_
    const long bh = j / (NBF_ * DH_);
    const long rem = j - bh * (NBF_ * DH_);
    float s = 0.f;
#pragma unroll
    for (int zk = 0; zk < KSPLIT_; ++zk)
        s += part[(bh * KSPLIT_ + zk) * (long)(NBF_ * DH_) + rem];
    ctx[j] = s;
}

// Batched f32 -> f16 transpose (R,C multiples of 32): out[z][c][r] = in[z][r][c]
__global__ void transpose_f32_to_f16(
    const float* __restrict__ in, _Float16* __restrict__ out, int R, int C)
{
    __shared__ float tile[32][33];
    const int z = blockIdx.z;
    const float* inz = in + (long)z * R * C;
    _Float16* outz = out + (long)z * R * C;
    const int c0 = blockIdx.x * 32, r0 = blockIdx.y * 32;
    const int tx = threadIdx.x, ty = threadIdx.y;
#pragma unroll
    for (int j = 0; j < 4; ++j)
        tile[ty + 8 * j][tx] = inz[(long)(r0 + ty + 8 * j) * C + c0 + tx];
    __syncthreads();
#pragma unroll
    for (int j = 0; j < 4; ++j)
        outz[(long)(c0 + ty + 8 * j) * R + r0 + tx] = (_Float16)tile[tx][ty + 8 * j];
}

// =====================================================================
extern "C" void kernel_launch(void* const* d_in, const int* in_sizes, int n_in,
                              void* d_out, int out_size, void* d_ws, size_t ws_size,
                              hipStream_t stream)
{
    (void)in_sizes; (void)n_in; (void)out_size; (void)ws_size;
    const float* x    = (const float*)d_in[0];
    const float* ln1g = (const float*)d_in[1];
    const float* ln1b = (const float*)d_in[2];
    const float* ln2g = (const float*)d_in[3];
    const float* ln2b = (const float*)d_in[4];
    const float* Wq = (const float*)d_in[5];
    const float* bq = (const float*)d_in[6];
    const float* Wk = (const float*)d_in[7];
    const float* bk = (const float*)d_in[8];
    const float* Wv = (const float*)d_in[9];
    const float* bv = (const float*)d_in[10];
    const float* Wo = (const float*)d_in[11];
    const float* bo = (const float*)d_in[12];
    const float* proj = (const float*)d_in[13];
    const float* W1 = (const float*)d_in[14];
    const float* b1 = (const float*)d_in[15];
    const float* W2 = (const float*)d_in[16];
    const float* b2 = (const float*)d_in[17];

    char* wsp = (char*)d_ws; size_t off = 0;
    auto carve = [&](size_t bytes) -> void* {
        void* p = wsp + off; off = (off + bytes + 255) & ~(size_t)255; return p;
    };
    float*    xbuf  = (float*)   carve((size_t)NT_ * DIM_ * 4);
    _Float16* y16   = (_Float16*)carve((size_t)NT_ * DIM_ * 2);
    float*    qf    = (float*)   carve((size_t)NT_ * DIM_ * 4);
    float*    kf    = (float*)   carve((size_t)NT_ * DIM_ * 4);
    _Float16* vT    = (_Float16*)carve((size_t)DIM_ * NT_ * 2);       // [B,H,DH,N]
    _Float16* kpT   = (_Float16*)carve((size_t)BH_ * NBF_ * N_ * 2);  // [B,H,M,N]
    _Float16* qp16  = (_Float16*)carve((size_t)BH_ * N_ * NBF_ * 2);  // [B,H,N,M]
    float*    ksum  = (float*)   carve((size_t)BH_ * NBF_ * 4);
    unsigned* kmaxb = (unsigned*)carve((size_t)BH_ * 4);
    float*    dinv  = (float*)   carve((size_t)BH_ * N_ * 4);
    float*    ctxp  = (float*)   carve((size_t)BH_ * KSPLIT_ * NBF_ * DH_ * 4);
    float*    ctxf  = (float*)   carve((size_t)BH_ * NBF_ * DH_ * 4);
    _Float16* ctxT  = (_Float16*)carve((size_t)BH_ * DH_ * NBF_ * 2);
    _Float16* o16   = (_Float16*)carve((size_t)BH_ * N_ * DH_ * 2);   // [B,H,N,DH]
    _Float16* h16   = (_Float16*)carve((size_t)NT_ * HID_ * 2);
    _Float16* WqT   = (_Float16*)carve((size_t)DIM_ * DIM_ * 2);
    _Float16* WkT   = (_Float16*)carve((size_t)DIM_ * DIM_ * 2);
    _Float16* WvT   = (_Float16*)carve((size_t)DIM_ * DIM_ * 2);
    _Float16* WoT   = (_Float16*)carve((size_t)DIM_ * DIM_ * 2);
    _Float16* W1T   = (_Float16*)carve((size_t)DIM_ * HID_ * 2);
    _Float16* W2T   = (_Float16*)carve((size_t)HID_ * DIM_ * 2);

    hipMemcpyAsync(xbuf, x, (size_t)NT_ * DIM_ * 4, hipMemcpyDeviceToDevice, stream);

    const dim3 tb(32, 8);
    for (int i = 0; i < 2; ++i) {
        const float* pWq = Wq + (size_t)i * DIM_ * DIM_;
        const float* pWk = Wk + (size_t)i * DIM_ * DIM_;
        const float* pWv = Wv + (size_t)i * DIM_ * DIM_;
        const float* pWo = Wo + (size_t)i * DIM_ * DIM_;
        const float* pW1 = W1 + (size_t)i * DIM_ * HID_;
        const float* pW2 = W2 + (size_t)i * HID_ * DIM_;
        const float* pproj = proj + (size_t)i * NBF_ * DH_;

        transpose_f32_to_f16<<<dim3(DIM_/32, DIM_/32, 1), tb, 0, stream>>>(pWq, WqT, DIM_, DIM_);
        transpose_f32_to_f16<<<dim3(DIM_/32, DIM_/32, 1), tb, 0, stream>>>(pWk, WkT, DIM_, DIM_);
        transpose_f32_to_f16<<<dim3(DIM_/32, DIM_/32, 1), tb, 0, stream>>>(pWv, WvT, DIM_, DIM_);
        transpose_f32_to_f16<<<dim3(DIM_/32, DIM_/32, 1), tb, 0, stream>>>(pWo, WoT, DIM_, DIM_);
        transpose_f32_to_f16<<<dim3(HID_/32, DIM_/32, 1), tb, 0, stream>>>(pW1, W1T, DIM_, HID_);
        transpose_f32_to_f16<<<dim3(DIM_/32, HID_/32, 1), tb, 0, stream>>>(pW2, W2T, HID_, DIM_);

        // --- attention branch ---
        ln_kernel<<<NT_, 256, 0, stream>>>(xbuf, ln1g + i*DIM_, ln1b + i*DIM_, y16);

        const dim3 gq(NT_/128, DIM_/64, 1);
        gemm_wmma<0,0,true,false,false,false><<<gq, 256, 0, stream>>>(
            y16, DIM_, 0, WqT, DIM_, 0, bq + i*DIM_, nullptr, 0, nullptr,
            qf, DIM_, 0, DIM_, 1);
        gemm_wmma<0,0,true,false,false,false><<<gq, 256, 0, stream>>>(
            y16, DIM_, 0, WkT, DIM_, 0, bk + i*DIM_, nullptr, 0, nullptr,
            kf, DIM_, 0, DIM_, 1);
        gemm_wmma<0,2,true,false,false,false><<<gq, 256, 0, stream>>>(
            y16, DIM_, 0, WvT, DIM_, 0, bv + i*DIM_, nullptr, 0, nullptr,
            vT, 0, 0, DIM_, 1);

        hipMemsetAsync(kmaxb, 0, BH_ * 4, stream);
        kmax_kernel<<<dim3(N_/KROWS, BH_), 256, 0, stream>>>(kf, pproj, kmaxb);
        kp_kernel  <<<dim3(N_/KROWS, BH_), 256, 0, stream>>>(kf, pproj, kmaxb, kpT);
        ksum_kernel<<<BH_ * NBF_, 256, 0, stream>>>(kpT, ksum);
        qp_kernel  <<<dim3(N_/QROWS, BH_), 256, 0, stream>>>(qf, pproj, ksum, qp16, dinv);

        // ctx[b,h] = kp^T @ v  (M=256,N=64,K=8192), deterministic split-K
        gemm_wmma<0,3,false,false,false,false><<<dim3(NBF_/128, 1, BH_*KSPLIT_), 256, 0, stream>>>(
            kpT, N_, (long)NBF_*N_, vT, N_, (long)DH_*N_, nullptr, nullptr, 0, nullptr,
            ctxp, DH_, (long)NBF_*DH_, N_, KSPLIT_);
        ctx_reduce_kernel<<<(BH_*NBF_*DH_)/256, 256, 0, stream>>>(ctxp, ctxf);
        transpose_f32_to_f16<<<dim3(DH_/32, NBF_/32, BH_), tb, 0, stream>>>(ctxf, ctxT, NBF_, DH_);

        // o[b,h] = (qp @ ctx) * d_inv  (M=8192,N=64,K=256)
        gemm_wmma<0,1,false,false,true,false><<<dim3(N_/128, 1, BH_), 256, 0, stream>>>(
            qp16, NBF_, (long)N_*NBF_, ctxT, NBF_, (long)DH_*NBF_, nullptr,
            dinv, N_, nullptr, o16, DH_, (long)N_*DH_, NBF_, 1);

        // x += o @ Wo + bo   (A read in head-blocked layout)
        gemm_wmma<1,0,true,false,false,true><<<dim3(NT_/128, DIM_/64, 1), 256, 0, stream>>>(
            o16, 0, 0, WoT, DIM_, 0, bo + i*DIM_, nullptr, 0, xbuf,
            xbuf, DIM_, 0, DIM_, 1);

        // --- FFN branch ---
        ln_kernel<<<NT_, 256, 0, stream>>>(xbuf, ln2g + i*DIM_, ln2b + i*DIM_, y16);

        gemm_wmma<0,1,true,true,false,false><<<dim3(NT_/128, HID_/64, 1), 256, 0, stream>>>(
            y16, DIM_, 0, W1T, DIM_, 0, b1 + i*HID_, nullptr, 0, nullptr,
            h16, HID_, 0, DIM_, 1);

        float* outx = (i == 1) ? (float*)d_out : xbuf;
        gemm_wmma<0,0,true,false,false,true><<<dim3(NT_/128, DIM_/64, 1), 256, 0, stream>>>(
            h16, HID_, 0, W2T, HID_, 0, b2 + i*DIM_, nullptr, 0, xbuf,
            outx, DIM_, 0, HID_, 1);
    }
}
